// EnrichedHierarchicalGraphModule_14705968021780
// MI455X (gfx1250) — compile-verified, matrix operand order checked by reading, not run on hardware
//
#include <hip/hip_runtime.h>
#include <hip/hip_bf16.h>

#define N_USERS 20000
#define N_ITEMS 30000
#define NTOT    50000
#define DIM     64
#define NEDGE   500000
#define NCLUS   100
#define ORD_NEG_INF 0x007FFFFFu   // ordered-uint encoding of -inf

typedef __attribute__((ext_vector_type(16))) _Float16 v16h;
typedef __attribute__((ext_vector_type(8)))  float    v8f;

__device__ __forceinline__ unsigned fordf(float f) {
  unsigned u = __float_as_uint(f);
  return (u & 0x80000000u) ? ~u : (u | 0x80000000u);
}
__device__ __forceinline__ float funordf(unsigned u) {
  return __uint_as_float((u & 0x80000000u) ? (u & 0x7fffffffu) : ~u);
}

// ---------------- fills / copies ----------------
__global__ void k_fill_f32(float* p, float v, long long n) {
  long long i = (long long)blockIdx.x * blockDim.x + threadIdx.x;
  if (i < n) p[i] = v;
}
__global__ void k_fill_u32(unsigned* p, unsigned v, long long n) {
  long long i = (long long)blockIdx.x * blockDim.x + threadIdx.x;
  if (i < n) p[i] = v;
}
__global__ void k_copy_f32(float* d, const float* s, long long n) {
  long long i = (long long)blockIdx.x * blockDim.x + threadIdx.x;
  if (i < n) d[i] = s[i];
}

// ---------------- WMMA GEMM ----------------
// C[M,N] = act( concat(A1[M,K1], A2[M,K2]) @ W[K1+K2, N] + bias )
// f32 in memory, f16 WMMA math, f32 accumulate. One 16x16 C tile per wave.
// Requirements: N % 16 == 0, K1 % 32 == 0 (or K1 == total K), (K1+K2) % 32 == 0.
__launch_bounds__(256)
__global__ void k_wmma_gemm(const float* A1, int K1, const float* A2, int K2,
                            const float* __restrict__ W,
                            const float* __restrict__ bias,
                            float* __restrict__ C,
                            int M, int N, int act) {
  const int K = K1 + K2;
  const int tilesN = N >> 4;
  const int tilesM = (M + 15) >> 4;
  const int tile = blockIdx.x * (blockDim.x >> 5) + (threadIdx.x >> 5);
  if (tile >= tilesM * tilesN) return;
  const int tm = tile / tilesN;
  const int tn = tile - tm * tilesN;
  const int lane = threadIdx.x & 31;
  const int half = lane >> 4;          // 0: lanes 0-15, 1: lanes 16-31
  const int mr   = lane & 15;
  const int m = tm * 16 + mr;          // A row for this lane
  const int n = tn * 16 + mr;          // B/C column for this lane

  union { v8f v; float e[8]; } acc;
#pragma unroll
  for (int r = 0; r < 8; ++r) acc.e[r] = 0.0f;

  for (int k0 = 0; k0 < K; k0 += 32) {
    union { v16h v; _Float16 e[16]; } af, bf;
    // ---- A fragment: 16x32 f16 layout, lane(m, half), elems 0-7 -> K=k0+8h..,
    //      elems 8-15 -> K=k0+16+8h..  (two contiguous 32B runs -> b128 loads)
    const float* arow; int kk;
    if (k0 < K1) { arow = A1 + (size_t)m * K1; kk = k0; }
    else         { arow = A2 + (size_t)m * K2; kk = k0 - K1; }
    if (m < M) {
      const float4* p0 = reinterpret_cast<const float4*>(arow + kk + half * 8);
      const float4* p1 = reinterpret_cast<const float4*>(arow + kk + 16 + half * 8);
      float4 x0 = p0[0], x1 = p0[1], y0 = p1[0], y1 = p1[1];
      af.e[0]=(_Float16)x0.x; af.e[1]=(_Float16)x0.y; af.e[2]=(_Float16)x0.z; af.e[3]=(_Float16)x0.w;
      af.e[4]=(_Float16)x1.x; af.e[5]=(_Float16)x1.y; af.e[6]=(_Float16)x1.z; af.e[7]=(_Float16)x1.w;
      af.e[8]=(_Float16)y0.x; af.e[9]=(_Float16)y0.y; af.e[10]=(_Float16)y0.z; af.e[11]=(_Float16)y0.w;
      af.e[12]=(_Float16)y1.x; af.e[13]=(_Float16)y1.y; af.e[14]=(_Float16)y1.z; af.e[15]=(_Float16)y1.w;
    } else {
#pragma unroll
      for (int i = 0; i < 16; ++i) af.e[i] = (_Float16)0.0f;
    }
    // ---- B fragment: 32x16, same K striping, column n fixed per lane
#pragma unroll
    for (int i = 0; i < 16; ++i) {
      int k = k0 + half * 8 + (i & 7) + ((i >> 3) << 4);
      bf.e[i] = (_Float16)W[(size_t)k * N + n];
    }
    acc.v = __builtin_amdgcn_wmma_f32_16x16x32_f16(
        false, af.v, false, bf.v, (short)0, acc.v, false, false);
  }
  const float bn = bias ? bias[n] : 0.0f;
#pragma unroll
  for (int r = 0; r < 8; ++r) {
    int mo = tm * 16 + half * 8 + r;   // C VGPR r: M = r (+8 for upper half-lanes)
    if (mo < M) {
      float v = acc.e[r] + bn;
      if (act) v = fmaxf(v, 0.0f);
      C[(size_t)mo * N + n] = v;
    }
  }
}

// ---------------- GCN ----------------
__global__ void k_gcn_deg(const int* __restrict__ ei, const int* __restrict__ ca,
                          float* deg, int* clcnt, int E) {
  int e = blockIdx.x * blockDim.x + threadIdx.x;
  if (e >= E) return;
  int s = ei[e], d = ei[E + e];
  if (ca[s] == ca[d]) { atomicAdd(&deg[d], 1.0f); atomicAdd(&clcnt[ca[s]], 1); }
}
__global__ void k_rsqrt_ip(float* p, long long n) {
  long long i = (long long)blockIdx.x * blockDim.x + threadIdx.x;
  if (i < n) p[i] = rsqrtf(p[i]);
}
__global__ void k_gcn_self(const float* __restrict__ XW, const float* __restrict__ dinv,
                           float* out, long long n) {
  long long i = (long long)blockIdx.x * blockDim.x + threadIdx.x;
  if (i >= n) return;
  int nd = (int)(i >> 6);
  out[i] = XW[i] * dinv[nd] * dinv[nd];
}
__global__ void k_gcn_scatter(const int* __restrict__ ei, const int* __restrict__ ca,
                              const float* __restrict__ XW, const float* __restrict__ dinv,
                              float* out, int E) {
  long long t = (long long)blockIdx.x * blockDim.x + threadIdx.x;
  if (t >= (long long)E * DIM) return;
  int e = (int)(t >> 6), d = (int)(t & 63);
  int s = ei[e], dt = ei[E + e];
  if (ca[s] == ca[dt])
    atomicAdd(&out[(size_t)dt * DIM + d], XW[(size_t)s * DIM + d] * dinv[s] * dinv[dt]);
}
__global__ void k_gcn_select(const float* __restrict__ gout, const float* __restrict__ X,
                             const float* __restrict__ gb, const int* __restrict__ ca,
                             const int* __restrict__ clcnt, float* Xc, long long n) {
  long long i = (long long)blockIdx.x * blockDim.x + threadIdx.x;
  if (i >= n) return;
  int nd = (int)(i >> 6), d = (int)(i & 63);
  Xc[i] = (clcnt[ca[nd]] > 0) ? (gout[i] + gb[d]) : X[i];
}

// ---------------- cluster mean-pool (LDS staged) ----------------
__global__ void k_pool(const float* __restrict__ Xc, const int* __restrict__ ca,
                       float* psum, float* pcnt, int nnodes) {
  __shared__ float ls[NCLUS * DIM];
  __shared__ float lc[NCLUS];
  for (int i = threadIdx.x; i < NCLUS * DIM; i += blockDim.x) ls[i] = 0.0f;
  for (int i = threadIdx.x; i < NCLUS; i += blockDim.x) lc[i] = 0.0f;
  __syncthreads();
  long long total = (long long)nnodes * DIM;
  for (long long t = (long long)blockIdx.x * blockDim.x + threadIdx.x; t < total;
       t += (long long)gridDim.x * blockDim.x) {
    int nd = (int)(t >> 6), d = (int)(t & 63);
    int c = ca[nd];
    atomicAdd(&ls[c * DIM + d], Xc[t]);
    if (d == 0) atomicAdd(&lc[c], 1.0f);
  }
  __syncthreads();
  for (int i = threadIdx.x; i < NCLUS * DIM; i += blockDim.x)
    if (ls[i] != 0.0f) atomicAdd(&psum[i], ls[i]);
  for (int i = threadIdx.x; i < NCLUS; i += blockDim.x)
    if (lc[i] != 0.0f) atomicAdd(&pcnt[i], lc[i]);
}
__global__ void k_pool_div(const float* psum, const float* pcnt, float* pooled) {
  int i = blockIdx.x * blockDim.x + threadIdx.x;
  if (i >= NCLUS * DIM) return;
  pooled[i] = psum[i] / fmaxf(pcnt[i >> 6], 1.0f);
}

// ---------------- GAT pieces ----------------
// per-node attention logits: asrc[n,h] = <xh[n,h,:], att_src[h,:]>
__global__ void k_attn_node(const float* __restrict__ xh, const float* __restrict__ as_,
                            const float* __restrict__ ad_, float* asrc, float* adst,
                            int M, int H) {
  int t = blockIdx.x * blockDim.x + threadIdx.x;
  if (t >= M * H) return;
  int n = t / H, h = t - n * H;
  const float* xr = xh + (size_t)n * H * DIM + h * DIM;
  const float* a1 = as_ + h * DIM;
  const float* a2 = ad_ + h * DIM;
  float s1 = 0.f, s2 = 0.f;
  for (int c = 0; c < DIM; ++c) { s1 += xr[c] * a1[c]; s2 += xr[c] * a2[c]; }
  asrc[t] = s1; adst[t] = s2;
}
// q[d,h] = sum_c lin_e[d, h*D+c] * att_e[h,c]  (ED=2)
__global__ void k_edge_q(const float* __restrict__ lin_e, const float* __restrict__ att_e,
                         float* q, int H) {
  int t = threadIdx.x;
  if (t >= 2 * H) return;
  int d = t / H, h = t - d * H;
  float s = 0.f;
  for (int c = 0; c < DIM; ++c) s += lin_e[(size_t)d * H * DIM + h * DIM + c] * att_e[h * DIM + c];
  q[d * H + h] = s;
}
// edge-attr mean (ED=2)
__global__ void k_attr_sum(const float* __restrict__ ea, float* out2, int E) {
  float s0 = 0.f, s1 = 0.f;
  for (int e = blockIdx.x * blockDim.x + threadIdx.x; e < E; e += gridDim.x * blockDim.x) {
    s0 += ea[2 * e]; s1 += ea[2 * e + 1];
  }
  atomicAdd(&out2[0], s0); atomicAdd(&out2[1], s1);
}
__global__ void k_scale2(float* v, float s) { if (threadIdx.x < 2) v[threadIdx.x] *= s; }

// pass1: leaky-relu logits + segment max (ordered-uint atomicMax). idx>=E are self loops.
__global__ void k_edge_p1(const int* __restrict__ ei, const float* __restrict__ ea,
                          const float* __restrict__ eamean, const float* __restrict__ q,
                          const float* __restrict__ asrc, const float* __restrict__ adst,
                          float* al, unsigned* amax, int E, int nLoop, int H) {
  int idx = blockIdx.x * blockDim.x + threadIdx.x;
  if (idx >= E + nLoop) return;
  int s, d; float e0, e1;
  if (idx < E) { s = ei[idx]; d = ei[E + idx]; e0 = ea[2 * idx]; e1 = ea[2 * idx + 1]; }
  else         { s = d = idx - E; e0 = eamean[0]; e1 = eamean[1]; }
  for (int h = 0; h < H; ++h) {
    float a = asrc[s * H + h] + adst[d * H + h] + e0 * q[h] + e1 * q[H + h];
    a = (a > 0.f) ? a : 0.2f * a;
    al[(size_t)idx * H + h] = a;
    atomicMax(&amax[d * H + h], fordf(a));
  }
}
// pass2: ex = exp(al - max), segment sum. Overwrites al with ex.
__global__ void k_edge_p2(const int* __restrict__ ei, float* al,
                          const unsigned* __restrict__ amax, float* ssum,
                          int E, int nLoop, int H) {
  int idx = blockIdx.x * blockDim.x + threadIdx.x;
  if (idx >= E + nLoop) return;
  int d = (idx < E) ? ei[E + idx] : (idx - E);
  for (int h = 0; h < H; ++h) {
    float ex = __expf(al[(size_t)idx * H + h] - funordf(amax[d * H + h]));
    al[(size_t)idx * H + h] = ex;
    atomicAdd(&ssum[d * H + h], ex);
  }
}
// pass3: out[d,h,:] += xh[s,h,:] * ex/(sum+eps)
__global__ void k_edge_p3(const int* __restrict__ ei, const float* __restrict__ ex,
                          const float* __restrict__ ssum, const float* __restrict__ xh,
                          float* out, int E, int nLoop, int H) {
  long long t = (long long)blockIdx.x * blockDim.x + threadIdx.x;
  long long total = (long long)(E + nLoop) * H * DIM;
  if (t >= total) return;
  int c = (int)(t & 63);
  long long r = t >> 6;
  int h = (int)(r % H);
  int idx = (int)(r / H);
  int s, d;
  if (idx < E) { s = ei[idx]; d = ei[E + idx]; } else { s = d = idx - E; }
  float w = ex[(size_t)idx * H + h] / (ssum[d * H + h] + 1e-16f);
  atomicAdd(&out[(size_t)d * H * DIM + h * DIM + c],
            xh[(size_t)s * H * DIM + h * DIM + c] * w);
}
__global__ void k_head_mean(const float* __restrict__ aggr, const float* __restrict__ bias,
                            float* out, long long rows, int H) {
  long long i = (long long)blockIdx.x * blockDim.x + threadIdx.x;
  if (i >= rows * DIM) return;
  long long n = i >> 6; int d = (int)(i & 63);
  float s = 0.f;
  for (int h = 0; h < H; ++h) s += aggr[(size_t)n * H * DIM + h * DIM + d];
  out[i] = s * (1.0f / H) + bias[d];
}
__global__ void k_add_cluster(const float* __restrict__ Xc, const float* __restrict__ clupd,
                              const int* __restrict__ ca, float* Xcomb, long long n) {
  long long i = (long long)blockIdx.x * blockDim.x + threadIdx.x;
  if (i >= n) return;
  int nd = (int)(i >> 6), d = (int)(i & 63);
  Xcomb[i] = Xc[i] + clupd[(size_t)ca[nd] * DIM + d];
}

// ---------------- GraphNorm ----------------
__global__ void k_colreduce(const float* __restrict__ g, const float* __restrict__ mean,
                            const float* __restrict__ ms, float* out, long long rows, int mode) {
  __shared__ float ls[DIM];
  if (threadIdx.x < DIM) ls[threadIdx.x] = 0.0f;
  __syncthreads();
  long long total = rows * DIM;
  for (long long t = (long long)blockIdx.x * blockDim.x + threadIdx.x; t < total;
       t += (long long)gridDim.x * blockDim.x) {
    int d = (int)(t & 63);
    float v = g[t];
    if (mode) { v -= ms[d] * mean[d]; v *= v; }
    atomicAdd(&ls[d], v);
  }
  __syncthreads();
  if (threadIdx.x < DIM) atomicAdd(&out[threadIdx.x], ls[threadIdx.x]);
}
__global__ void k_colscale(float* p, float s) { if (threadIdx.x < DIM) p[threadIdx.x] *= s; }
__global__ void k_gn_final(const float* __restrict__ g, const float* __restrict__ mean,
                           const float* __restrict__ var, const float* __restrict__ w,
                           const float* __restrict__ b, const float* __restrict__ ms,
                           const float* __restrict__ Xin, float* out, long long rows) {
  long long i = (long long)blockIdx.x * blockDim.x + threadIdx.x;
  if (i >= rows * DIM) return;
  int d = (int)(i & 63);
  float v = g[i] - ms[d] * mean[d];
  v = w[d] * v * rsqrtf(var[d] + 1e-5f) + b[d] + Xin[i];
  out[i] = (v > 0.f) ? v : expm1f(v);
}

// ---------------- host ----------------
static inline int cdiv_i(long long a, int b) { return (int)((a + b - 1) / b); }

extern "C" void kernel_launch(void* const* d_in, const int* in_sizes, int n_in,
                              void* d_out, int out_size, void* d_ws, size_t ws_size,
                              hipStream_t stream) {
  (void)n_in; (void)out_size; (void)ws_size;
  const float* extra    = (const float*)d_in[0];
  const float* user_emb = (const float*)d_in[1];
  const float* item_emb = (const float*)d_in[2];
  const float* fw1 = (const float*)d_in[3]; const float* fb1 = (const float*)d_in[4];
  const float* fw2 = (const float*)d_in[5]; const float* fb2 = (const float*)d_in[6];
  const float* fea  = (const float*)d_in[7];
  const float* gcnw = (const float*)d_in[8]; const float* gcnb = (const float*)d_in[9];
  const float* cg_lin = (const float*)d_in[10];
  const float* cg_as = (const float*)d_in[11]; const float* cg_ad = (const float*)d_in[12];
  const float* cg_le = (const float*)d_in[13]; const float* cg_ae = (const float*)d_in[14];
  const float* cg_b  = (const float*)d_in[15];
  const float* cea   = (const float*)d_in[16];
  const float* g_lin = (const float*)d_in[17];
  const float* g_as  = (const float*)d_in[18]; const float* g_ad = (const float*)d_in[19];
  const float* g_le  = (const float*)d_in[20]; const float* g_ae = (const float*)d_in[21];
  const float* g_b   = (const float*)d_in[22];
  const float* gn_w  = (const float*)d_in[23]; const float* gn_bi = (const float*)d_in[24];
  const float* gn_ms = (const float*)d_in[25];
  const int* fei = (const int*)d_in[26];
  const int* ca  = (const int*)d_in[27];
  const int* cei = (const int*)d_in[28];
  const int MC = in_sizes[16] / 2;   // #inter-cluster edges

  char* wp = (char*)d_ws;
  auto alloc = [&](size_t bytes) -> void* {
    void* p = (void*)wp; wp += (bytes + 255) & ~(size_t)255; return p;
  };
  float*    h1     = (float*)alloc((size_t)N_ITEMS * 128 * 4);
  float*    X      = (float*)alloc((size_t)NTOT * DIM * 4);
  float*    XW     = (float*)alloc((size_t)NTOT * DIM * 4);
  float*    gout   = (float*)alloc((size_t)NTOT * DIM * 4);
  float*    Xc     = (float*)alloc((size_t)NTOT * DIM * 4);
  float*    XA     = (float*)alloc((size_t)NTOT * DIM * 4);
  float*    XB     = (float*)alloc((size_t)NTOT * DIM * 4);
  float*    xh     = (float*)alloc((size_t)NTOT * 128 * 4);
  float*    aggr   = (float*)alloc((size_t)NTOT * 128 * 4);
  float*    gbuf   = (float*)alloc((size_t)NTOT * DIM * 4);
  float*    deg    = (float*)alloc((size_t)NTOT * 4);
  int*      clcnt  = (int*)alloc(NCLUS * 4);
  float*    psum   = (float*)alloc(NCLUS * DIM * 4);
  float*    pcnt   = (float*)alloc(NCLUS * 4);
  float*    pooled = (float*)alloc(NCLUS * DIM * 4);
  float*    xh_c   = (float*)alloc(NCLUS * 256 * 4);
  float*    asrc_c = (float*)alloc(NCLUS * 4 * 4);
  float*    adst_c = (float*)alloc(NCLUS * 4 * 4);
  float*    q_c    = (float*)alloc(8 * 4);
  unsigned* amax_c = (unsigned*)alloc(NCLUS * 4 * 4);
  float*    ssum_c = (float*)alloc(NCLUS * 4 * 4);
  float*    al_c   = (float*)alloc((size_t)(MC + NCLUS) * 4 * 4);
  float*    aggr_c = (float*)alloc(NCLUS * 256 * 4);
  float*    clupd  = (float*)alloc(NCLUS * DIM * 4);
  float*    ceam   = (float*)alloc(2 * 4);
  float*    feam   = (float*)alloc(2 * 4);
  float*    qf     = (float*)alloc(4 * 4);
  float*    asrc   = (float*)alloc((size_t)NTOT * 2 * 4);
  float*    adst   = (float*)alloc((size_t)NTOT * 2 * 4);
  unsigned* amax   = (unsigned*)alloc((size_t)NTOT * 2 * 4);
  float*    ssum   = (float*)alloc((size_t)NTOT * 2 * 4);
  float*    al     = (float*)alloc((size_t)(NEDGE + NTOT) * 2 * 4);
  float*    colsum = (float*)alloc(DIM * 4);
  float*    colsq  = (float*)alloc(DIM * 4);

  const int B = 256;
  auto gemm = [&](const float* A1, int K1, const float* A2, int K2, const float* W,
                  const float* bias, float* C, int M, int N, int act) {
    int tiles = ((M + 15) >> 4) * (N >> 4);
    k_wmma_gemm<<<cdiv_i(tiles, 8), B, 0, stream>>>(A1, K1, A2, K2, W, bias, C, M, N, act);
  };

  // ---- fusion MLP (item side) + assemble X ----
  gemm(item_emb, DIM, extra, 1152, fw1, fb1, h1, N_ITEMS, 128, 1);
  gemm(h1, 128, h1, 0, fw2, fb2, X + (size_t)N_USERS * DIM, N_ITEMS, DIM, 0);
  k_copy_f32<<<cdiv_i((long long)N_USERS * DIM, B), B, 0, stream>>>(X, user_emb, (long long)N_USERS * DIM);

  // ---- intra-cluster GCN ----
  k_fill_f32<<<cdiv_i(NTOT, B), B, 0, stream>>>(deg, 1.0f, NTOT);           // self-loop weight
  k_fill_u32<<<1, B, 0, stream>>>((unsigned*)clcnt, 0u, NCLUS);
  k_gcn_deg<<<cdiv_i(NEDGE, B), B, 0, stream>>>(fei, ca, deg, clcnt, NEDGE);
  k_rsqrt_ip<<<cdiv_i(NTOT, B), B, 0, stream>>>(deg, NTOT);                 // deg -> dinv
  gemm(X, DIM, X, 0, gcnw, nullptr, XW, NTOT, DIM, 0);
  k_gcn_self<<<cdiv_i((long long)NTOT * DIM, B), B, 0, stream>>>(XW, deg, gout, (long long)NTOT * DIM);
  k_gcn_scatter<<<cdiv_i((long long)NEDGE * DIM, B), B, 0, stream>>>(fei, ca, XW, deg, gout, NEDGE);
  k_gcn_select<<<cdiv_i((long long)NTOT * DIM, B), B, 0, stream>>>(gout, X, gcnb, ca, clcnt, Xc, (long long)NTOT * DIM);

  // ---- cluster mean pool ----
  k_fill_f32<<<cdiv_i(NCLUS * DIM, B), B, 0, stream>>>(psum, 0.0f, NCLUS * DIM);
  k_fill_f32<<<1, B, 0, stream>>>(pcnt, 0.0f, NCLUS);
  k_pool<<<512, B, 0, stream>>>(Xc, ca, psum, pcnt, NTOT);
  k_pool_div<<<cdiv_i(NCLUS * DIM, B), B, 0, stream>>>(psum, pcnt, pooled);

  // ---- global GAT over clusters (H=4, with self loops) ----
  gemm(pooled, DIM, pooled, 0, cg_lin, nullptr, xh_c, NCLUS, 256, 0);
  k_attn_node<<<cdiv_i(NCLUS * 4, B), B, 0, stream>>>(xh_c, cg_as, cg_ad, asrc_c, adst_c, NCLUS, 4);
  k_edge_q<<<1, 64, 0, stream>>>(cg_le, cg_ae, q_c, 4);
  k_fill_f32<<<1, B, 0, stream>>>(ceam, 0.0f, 2);
  k_attr_sum<<<64, B, 0, stream>>>(cea, ceam, MC);
  k_scale2<<<1, 32, 0, stream>>>(ceam, 1.0f / (float)MC);
  k_fill_u32<<<cdiv_i(NCLUS * 4, B), B, 0, stream>>>(amax_c, ORD_NEG_INF, NCLUS * 4);
  k_fill_f32<<<cdiv_i(NCLUS * 4, B), B, 0, stream>>>(ssum_c, 0.0f, NCLUS * 4);
  k_edge_p1<<<cdiv_i(MC + NCLUS, B), B, 0, stream>>>(cei, cea, ceam, q_c, asrc_c, adst_c, al_c, amax_c, MC, NCLUS, 4);
  k_edge_p2<<<cdiv_i(MC + NCLUS, B), B, 0, stream>>>(cei, al_c, amax_c, ssum_c, MC, NCLUS, 4);
  k_fill_f32<<<cdiv_i(NCLUS * 256, B), B, 0, stream>>>(aggr_c, 0.0f, NCLUS * 256);
  k_edge_p3<<<cdiv_i((long long)(MC + NCLUS) * 4 * DIM, B), B, 0, stream>>>(cei, al_c, ssum_c, xh_c, aggr_c, MC, NCLUS, 4);
  k_head_mean<<<cdiv_i(NCLUS * DIM, B), B, 0, stream>>>(aggr_c, cg_b, clupd, NCLUS, 4);
  k_add_cluster<<<cdiv_i((long long)NTOT * DIM, B), B, 0, stream>>>(Xc, clupd, ca, XA, (long long)NTOT * DIM);

  // ---- full-graph edge-attr mean (for self loops) ----
  k_fill_f32<<<1, B, 0, stream>>>(feam, 0.0f, 2);
  k_attr_sum<<<128, B, 0, stream>>>(fea, feam, NEDGE);
  k_scale2<<<1, 32, 0, stream>>>(feam, 1.0f / (float)NEDGE);

  // ---- 2 full-graph GAT layers (H=2) + GraphNorm + residual ELU ----
  const int ETOT = NEDGE + NTOT;
  for (int i = 0; i < 2; ++i) {
    const float* Xin = (i == 0) ? XA : XB;
    float* Xout = (i == 0) ? XB : (float*)d_out;
    gemm(Xin, DIM, Xin, 0, g_lin + (size_t)i * DIM * 128, nullptr, xh, NTOT, 128, 0);
    k_attn_node<<<cdiv_i(NTOT * 2, B), B, 0, stream>>>(xh, g_as + i * 128, g_ad + i * 128, asrc, adst, NTOT, 2);
    k_edge_q<<<1, 64, 0, stream>>>(g_le + i * 256, g_ae + i * 128, qf, 2);
    k_fill_u32<<<cdiv_i(NTOT * 2, B), B, 0, stream>>>(amax, ORD_NEG_INF, (long long)NTOT * 2);
    k_fill_f32<<<cdiv_i(NTOT * 2, B), B, 0, stream>>>(ssum, 0.0f, (long long)NTOT * 2);
    k_edge_p1<<<cdiv_i(ETOT, B), B, 0, stream>>>(fei, fea, feam, qf, asrc, adst, al, amax, NEDGE, NTOT, 2);
    k_edge_p2<<<cdiv_i(ETOT, B), B, 0, stream>>>(fei, al, amax, ssum, NEDGE, NTOT, 2);
    k_fill_f32<<<cdiv_i((long long)NTOT * 128, B), B, 0, stream>>>(aggr, 0.0f, (long long)NTOT * 128);
    k_edge_p3<<<cdiv_i((long long)ETOT * 2 * DIM, B), B, 0, stream>>>(fei, al, ssum, xh, aggr, NEDGE, NTOT, 2);
    k_head_mean<<<cdiv_i((long long)NTOT * DIM, B), B, 0, stream>>>(aggr, g_b + i * DIM, gbuf, NTOT, 2);
    // GraphNorm column stats (LDS-staged)
    k_fill_f32<<<1, B, 0, stream>>>(colsum, 0.0f, DIM);
    k_colreduce<<<1024, B, 0, stream>>>(gbuf, colsum, gn_ms + i * DIM, colsum, NTOT, 0);
    k_colscale<<<1, 64, 0, stream>>>(colsum, 1.0f / (float)NTOT);        // -> mean
    k_fill_f32<<<1, B, 0, stream>>>(colsq, 0.0f, DIM);
    k_colreduce<<<1024, B, 0, stream>>>(gbuf, colsum, gn_ms + i * DIM, colsq, NTOT, 1);
    k_colscale<<<1, 64, 0, stream>>>(colsq, 1.0f / (float)NTOT);         // -> var
    k_gn_final<<<cdiv_i((long long)NTOT * DIM, B), B, 0, stream>>>(
        gbuf, colsum, colsq, gn_w + i * DIM, gn_bi + i * DIM, gn_ms + i * DIM, Xin, Xout, NTOT);
  }
}
